// DeformableTransformer_14276471292329
// MI455X (gfx1250) — compile-verified, matrix operand order checked by reading, not run on hardware
//
#include <hip/hip_runtime.h>
#include <hip/hip_bf16.h>

// ---------------------------------------------------------------------------
// Problem constants (from the reference)
// ---------------------------------------------------------------------------
#define D_MODEL 256
#define NH      8
#define NL      4
#define NP      4
#define DH      32
#define DFF     1024
#define LEN_TOK 13294
#define LN_EPS  1e-5f

typedef __attribute__((ext_vector_type(16))) __bf16 v16bf;
typedef __attribute__((ext_vector_type(8)))  float  v8f;

__device__ __forceinline__ unsigned short f32_to_bf16(float f) {
    unsigned u = __float_as_uint(f);
    u += 0x7FFFu + ((u >> 16) & 1u);      // round to nearest even
    return (unsigned short)(u >> 16);
}

// CDNA5 async global->LDS copy (16B per lane), tracked by ASYNCcnt.
// lds_off is the wave-relative LDS byte offset = low 32 bits of the flat
// pointer into the LDS aperture.
__device__ __forceinline__ void async_load_b128(const void* gptr, unsigned lds_off) {
    asm volatile("global_load_async_to_lds_b128 %0, %1, off"
                 :: "v"(lds_off), "v"(gptr)
                 : "memory");
}

// ---------------------------------------------------------------------------
// 1. prep: query = src + pos ; bf16 copies of src and query
// ---------------------------------------------------------------------------
__global__ __launch_bounds__(256) void prep_kernel(
    const float* __restrict__ src, const float* __restrict__ pos,
    unsigned short* __restrict__ src_bf, unsigned short* __restrict__ q_bf,
    long long n)
{
    long long i = (long long)blockIdx.x * 256 + threadIdx.x;
    if (i >= n) return;
    float s = src[i];
    src_bf[i] = f32_to_bf16(s);
    q_bf[i]   = f32_to_bf16(s + pos[i]);
}

// ---------------------------------------------------------------------------
// 2. weight convert + transpose: f32 [K][N] -> bf16 [N][K]
// ---------------------------------------------------------------------------
__global__ __launch_bounds__(256) void conv_weight_kernel(
    const float* __restrict__ w, unsigned short* __restrict__ wt, int K, int N)
{
    int i = blockIdx.x * 256 + threadIdx.x;
    if (i >= K * N) return;
    int k = i / N, n = i % N;
    wt[(long long)n * K + k] = f32_to_bf16(w[i]);
}

// ---------------------------------------------------------------------------
// 3. WMMA GEMM:  C[M,N] = A[M,K](bf16) * Wt[N,K](bf16)^T + bias, opt ReLU,
//    output f32 and/or bf16.  128x64 tile / 256-thread block / 8 waves,
//    each wave a 32x32 quadrant = 2x2 v_wmma_f32_16x16x32_bf16 accumulators.
//    K-tiles are double-buffered through LDS with async global->LDS loads:
//    issue next tile, s_wait_asynccnt for the current one, compute, repeat.
// ---------------------------------------------------------------------------
#define TM 128
#define TN 64
#define TK 32
#define ASTR 40   // padded halfword stride (80B rows -> conflict-free b32 reads)
#define BSTR 40

__global__ __launch_bounds__(256) void gemm_wmma_kernel(
    const unsigned short* __restrict__ A,   // M x K  (row-major bf16)
    const unsigned short* __restrict__ Wt,  // N x K  (row-major bf16, pre-transposed)
    const float* __restrict__ bias,         // N
    float* __restrict__ outF,               // may be null
    unsigned short* __restrict__ outBF,     // may be null
    int M, int N, int K, int relu)
{
    __shared__ unsigned short As[2][TM * ASTR];
    __shared__ unsigned short Bs[2][TN * BSTR];

    const int tid  = threadIdx.x;
    const int lane = tid & 31;
    const int wave = tid >> 5;
    const int mb   = blockIdx.y * TM;
    const int nb   = blockIdx.x * TN;
    const int wm   = (wave >> 1) * 32;      // wave's M offset in tile
    const int wn   = (wave & 1) * 32;       // wave's N offset in tile
    const int row16 = lane & 15;
    const int half  = lane >> 4;
    const int h8    = half * 8;

    // per-thread staging coordinates
    const int ar = tid >> 1;               // A tile row 0..127
    const int ac = (tid & 1) * 16;         // A tile col 0 / 16
    int agm = mb + ar; if (agm >= M) agm = M - 1;
    const unsigned short* aptr = A + (long long)agm * K + ac;
    const int br = tid >> 2;               // B tile row 0..63
    const int bc = (tid & 3) * 8;          // B tile col
    const unsigned short* bptr = Wt + (long long)(nb + br) * K + bc;

    auto stage = [&](int k0, int buf) {
        unsigned la = (unsigned)(unsigned long long)&As[buf][ar * ASTR + ac];
        const unsigned short* sa = aptr + k0;
        async_load_b128(sa,     la);        // 16B
        async_load_b128(sa + 8, la + 16);   // next 16B
        unsigned lb = (unsigned)(unsigned long long)&Bs[buf][br * BSTR + bc];
        async_load_b128(bptr + k0, lb);
    };

    v8f acc[2][2] = {};
    const int niter = K / TK;

    stage(0, 0);
    for (int i = 0; i < niter; ++i) {
        int cur = i & 1;
        if (i + 1 < niter) {
            stage((i + 1) * TK, cur ^ 1);
            asm volatile("s_wait_asynccnt 0x3" ::: "memory"); // current buf done
        } else {
            asm volatile("s_wait_asynccnt 0x0" ::: "memory");
        }
        __syncthreads();

        // ---- build fragments per ISA 16-bit A/B VGPR layout ----
        union Frag { unsigned u[8]; v16bf v; };
        Frag af[2], bf[2];
#pragma unroll
        for (int ii = 0; ii < 2; ++ii) {
            int m = wm + ii * 16 + row16;
#pragma unroll
            for (int r = 0; r < 8; ++r) {
                int k = ((r & 4) << 2) + h8 + ((r & 3) << 1); // K pair index
                af[ii].u[r] = *(const unsigned*)&As[cur][m * ASTR + k];
            }
        }
#pragma unroll
        for (int j = 0; j < 2; ++j) {
            int n = wn + j * 16 + row16;
#pragma unroll
            for (int r = 0; r < 8; ++r) {
                int k = ((r & 4) << 2) + h8 + ((r & 3) << 1);
                bf[j].u[r] = *(const unsigned*)&Bs[cur][n * BSTR + k];
            }
        }
#pragma unroll
        for (int ii = 0; ii < 2; ++ii)
#pragma unroll
            for (int j = 0; j < 2; ++j)
                acc[ii][j] = __builtin_amdgcn_wmma_f32_16x16x32_bf16(
                    false, af[ii].v, false, bf[j].v,
                    (short)0, acc[ii][j], false, false);

        __syncthreads();  // all waves done reading buf before it is re-filled
    }

    // ---- epilogue: bias (+ReLU) then f32 / bf16 stores ----
#pragma unroll
    for (int i = 0; i < 2; ++i) {
#pragma unroll
        for (int j = 0; j < 2; ++j) {
            int n = nb + wn + j * 16 + row16;
            float bv = bias[n];
#pragma unroll
            for (int r = 0; r < 8; ++r) {
                int m = mb + wm + i * 16 + h8 + r;   // C/D layout: M = half*8 + vgpr
                if (m < M) {
                    float v = acc[i][j][r] + bv;
                    if (relu) v = fmaxf(v, 0.f);
                    long long idx = (long long)m * N + n;
                    if (outF)  outF[idx]  = v;
                    if (outBF) outBF[idx] = f32_to_bf16(v);
                }
            }
        }
    }
}

// ---------------------------------------------------------------------------
// 4. softmax over the 16 (NL*NP) attention logits per (token, head)
// ---------------------------------------------------------------------------
__global__ __launch_bounds__(256) void softmax16_kernel(float* attn, int total)
{
    int i = blockIdx.x * 256 + threadIdx.x;
    if (i >= total) return;
    int row = i >> 3;          // token index
    int h   = i & 7;           // head
    float* p = attn + (long long)row * (NH * NL * NP) + h * (NL * NP);
    float mx = -1e30f;
#pragma unroll
    for (int j = 0; j < 16; ++j) mx = fmaxf(mx, p[j]);
    float e[16], s = 0.f;
#pragma unroll
    for (int j = 0; j < 16; ++j) { e[j] = __expf(p[j] - mx); s += e[j]; }
    float inv = 1.f / s;
#pragma unroll
    for (int j = 0; j < 16; ++j) p[j] = e[j] * inv;
}

// ---------------------------------------------------------------------------
// 5. deformable sampling: one wave per (token, head), lane = channel (DH=32)
// ---------------------------------------------------------------------------
__global__ __launch_bounds__(256) void deform_sample_kernel(
    const float* __restrict__ value,   // [B*LEN, 256]   (NH*DH channels)
    const float* __restrict__ off,     // [B*LEN, 256]   (NH,NL,NP,2)
    const float* __restrict__ attw,    // [B*LEN, 128]   (NH,NL,NP) post-softmax
    const float* __restrict__ ref,     // [B*LEN, NL, 2]
    unsigned short* __restrict__ samp_bf, // [B*LEN, 256] bf16
    int M, int len)
{
    const int lvlH[NL] = {100, 50, 25, 13};
    const int lvlW[NL] = {100, 50, 25, 13};
    const int lvlS[NL] = {0, 10000, 12500, 13125};

    int unit = blockIdx.x * 8 + (threadIdx.x >> 5);
    int lane = threadIdx.x & 31;               // channel within head
    if (unit >= M * NH) return;
    int bt = unit / NH;
    int h  = unit % NH;
    int b  = bt / len;

    const float* refp  = ref  + (long long)bt * NL * 2;
    const float* offp  = off  + (long long)bt * D_MODEL + h * NL * NP * 2;
    const float* awp   = attw + (long long)bt * (NH * NL * NP) + h * NL * NP;
    const float* vbase = value + (long long)b * len * D_MODEL + h * DH + lane;

    float acc = 0.f;
#pragma unroll
    for (int l = 0; l < NL; ++l) {
        const int Hs = lvlH[l], Ws = lvlW[l], S = lvlS[l];
        float rx = refp[l * 2 + 0], ry = refp[l * 2 + 1];
#pragma unroll
        for (int p = 0; p < NP; ++p) {
            float x = rx * (float)Ws + offp[(l * NP + p) * 2 + 0] - 0.5f;
            float y = ry * (float)Hs + offp[(l * NP + p) * 2 + 1] - 0.5f;
            float aw = awp[l * NP + p];
            float xf = floorf(x), yf = floorf(y);
            float dx = x - xf, dy = y - yf;
            int x0 = (int)xf, y0 = (int)yf;
            float w00 = (1.f - dx) * (1.f - dy) * aw;
            float w10 = dx * (1.f - dy) * aw;
            float w01 = (1.f - dx) * dy * aw;
            float w11 = dx * dy * aw;
#pragma unroll
            for (int c = 0; c < 4; ++c) {
                int xi = x0 + (c & 1);
                int yi = y0 + (c >> 1);
                float w = (c == 0) ? w00 : (c == 1) ? w10 : (c == 2) ? w01 : w11;
                if (xi >= 0 && xi < Ws && yi >= 0 && yi < Hs)
                    acc += w * vbase[(long long)(S + yi * Ws + xi) * D_MODEL];
            }
        }
    }
    samp_bf[(long long)bt * D_MODEL + h * DH + lane] = f32_to_bf16(acc);
}

// ---------------------------------------------------------------------------
// 6. fused residual add + LayerNorm (one token per 256-thread block)
// ---------------------------------------------------------------------------
__global__ __launch_bounds__(256) void add_layernorm_kernel(
    const float* __restrict__ a, const float* __restrict__ b,
    const float* __restrict__ g, const float* __restrict__ be,
    float* __restrict__ outF, unsigned short* __restrict__ outBF, int M)
{
    __shared__ float red[256];
    int t = blockIdx.x;
    if (t >= M) return;
    int c = threadIdx.x;
    long long idx = (long long)t * D_MODEL + c;
    float x = a[idx] + b[idx];

    red[c] = x; __syncthreads();
    for (int s = 128; s > 0; s >>= 1) { if (c < s) red[c] += red[c + s]; __syncthreads(); }
    float mean = red[0] * (1.f / D_MODEL);
    __syncthreads();

    float d = x - mean;
    red[c] = d * d; __syncthreads();
    for (int s = 128; s > 0; s >>= 1) { if (c < s) red[c] += red[c + s]; __syncthreads(); }
    float var = red[0] * (1.f / D_MODEL);

    float y = d * rsqrtf(var + LN_EPS) * g[c] + be[c];
    if (outF)  outF[idx]  = y;
    if (outBF) outBF[idx] = f32_to_bf16(y);
}

// ---------------------------------------------------------------------------
// launch
// ---------------------------------------------------------------------------
extern "C" void kernel_launch(void* const* d_in, const int* in_sizes, int n_in,
                              void* d_out, int out_size, void* d_ws, size_t ws_size,
                              hipStream_t stream)
{
    const float* src     = (const float*)d_in[0];
    const float* pos     = (const float*)d_in[1];
    const float* refpts  = (const float*)d_in[2];
    const float* W_value = (const float*)d_in[3];
    const float* b_value = (const float*)d_in[4];
    const float* W_off   = (const float*)d_in[5];
    const float* b_off   = (const float*)d_in[6];
    const float* W_attn  = (const float*)d_in[7];
    const float* b_attn  = (const float*)d_in[8];
    const float* W_out   = (const float*)d_in[9];
    const float* b_out   = (const float*)d_in[10];
    const float* W1      = (const float*)d_in[11];
    const float* b1      = (const float*)d_in[12];
    const float* W2      = (const float*)d_in[13];
    const float* b2      = (const float*)d_in[14];
    const float* g1      = (const float*)d_in[15];
    const float* be1     = (const float*)d_in[16];
    const float* g2      = (const float*)d_in[17];
    const float* be2     = (const float*)d_in[18];
    float* out           = (float*)d_out;
    (void)n_in; (void)out_size; (void)ws_size;

    const int M = in_sizes[0] / D_MODEL;          // B * LEN

    // ---- bump-allocate workspace (256B aligned) ----
    char* ws = (char*)d_ws;
    size_t off_b = 0;
    auto alloc = [&](size_t bytes) -> void* {
        void* p = ws + off_b;
        off_b = (off_b + bytes + 255) & ~(size_t)255;
        return p;
    };
    unsigned short* wv_t   = (unsigned short*)alloc((size_t)D_MODEL * D_MODEL * 2);
    unsigned short* woff_t = (unsigned short*)alloc((size_t)D_MODEL * D_MODEL * 2);
    unsigned short* wat_t  = (unsigned short*)alloc((size_t)D_MODEL * 128 * 2);
    unsigned short* wout_t = (unsigned short*)alloc((size_t)D_MODEL * D_MODEL * 2);
    unsigned short* w1_t   = (unsigned short*)alloc((size_t)D_MODEL * DFF * 2);
    unsigned short* w2_t   = (unsigned short*)alloc((size_t)DFF * D_MODEL * 2);
    unsigned short* src_bf = (unsigned short*)alloc((size_t)M * D_MODEL * 2);
    unsigned short* q_bf   = (unsigned short*)alloc((size_t)M * D_MODEL * 2);
    float* value           = (float*)alloc((size_t)M * D_MODEL * 4);
    float* offv            = (float*)alloc((size_t)M * D_MODEL * 4);
    float* attn            = (float*)alloc((size_t)M * 128 * 4);
    unsigned short* samp   = (unsigned short*)alloc((size_t)M * D_MODEL * 2);
    float* src2            = (float*)alloc((size_t)M * D_MODEL * 4);
    float* ln1             = (float*)alloc((size_t)M * D_MODEL * 4);
    unsigned short* ln1_bf = (unsigned short*)alloc((size_t)M * D_MODEL * 2);
    unsigned short* ffn_bf = (unsigned short*)alloc((size_t)M * DFF * 2);
    float* ffn2            = (float*)alloc((size_t)M * D_MODEL * 4);

    // ---- 1. prep ----
    long long nElem = (long long)M * D_MODEL;
    prep_kernel<<<(unsigned)((nElem + 255) / 256), 256, 0, stream>>>(src, pos, src_bf, q_bf, nElem);

    // ---- 2. weight convert/transpose ----
    auto convW = [&](const float* w, unsigned short* wt, int K, int N) {
        conv_weight_kernel<<<(K * N + 255) / 256, 256, 0, stream>>>(w, wt, K, N);
    };
    convW(W_value, wv_t,   D_MODEL, D_MODEL);
    convW(W_off,   woff_t, D_MODEL, D_MODEL);
    convW(W_attn,  wat_t,  D_MODEL, 128);
    convW(W_out,   wout_t, D_MODEL, D_MODEL);
    convW(W1,      w1_t,   D_MODEL, DFF);
    convW(W2,      w2_t,   DFF,     D_MODEL);

    const int mblk = (M + TM - 1) / TM;
    auto gemm = [&](const unsigned short* A, const unsigned short* Wt, const float* bias,
                    float* oF, unsigned short* oBF, int N, int K, int relu) {
        gemm_wmma_kernel<<<dim3(N / TN, mblk), 256, 0, stream>>>(A, Wt, bias, oF, oBF, M, N, K, relu);
    };

    // ---- 3. projections ----
    gemm(src_bf, wv_t,   b_value, value, nullptr, D_MODEL, D_MODEL, 0);
    gemm(q_bf,   woff_t, b_off,   offv,  nullptr, D_MODEL, D_MODEL, 0);
    gemm(q_bf,   wat_t,  b_attn,  attn,  nullptr, 128,     D_MODEL, 0);

    // ---- 4. softmax ----
    int totHT = M * NH;
    softmax16_kernel<<<(totHT + 255) / 256, 256, 0, stream>>>(attn, totHT);

    // ---- 5. deformable sampling ----
    deform_sample_kernel<<<(totHT + 7) / 8, 256, 0, stream>>>(value, offv, attn, refpts, samp, M, LEN_TOK);

    // ---- 6. output projection, residual + LN1 ----
    gemm(samp, wout_t, b_out, src2, nullptr, D_MODEL, D_MODEL, 0);
    add_layernorm_kernel<<<M, 256, 0, stream>>>(src, src2, g1, be1, ln1, ln1_bf, M);

    // ---- 7. FFN (ReLU fused into FFN1, bf16 intermediate) ----
    gemm(ln1_bf, w1_t, b1, nullptr, ffn_bf, DFF,     D_MODEL, 1);
    gemm(ffn_bf, w2_t, b2, ffn2,    nullptr, D_MODEL, DFF,    0);

    // ---- 8. residual + LN2 -> d_out ----
    add_layernorm_kernel<<<M, 256, 0, stream>>>(ln1, ffn2, g2, be2, out, nullptr, M);
}
